// My_Seg_49039936586288
// MI455X (gfx1250) — matrix operand order repaired and checked by reading, not compiled, not run on hardware
//
#include <hip/hip_runtime.h>
#include <hip/hip_bf16.h>
#include <math.h>

typedef __bf16 bf16_t;
typedef __bf16 v16bf __attribute__((ext_vector_type(16)));
typedef __bf16 v8bf  __attribute__((ext_vector_type(8)));
typedef float  v8f   __attribute__((ext_vector_type(8)));
typedef float  v4f   __attribute__((ext_vector_type(4)));

#define STATS_CH 1024
#define EPS 1e-5f

__device__ __forceinline__ float act_apply(float v, int act) {
  if (act == 1) return v > 0.f ? v : 0.f;            // relu
  if (act == 2) return v >= 0.f ? v : 0.2f * v;      // lrelu(0.2)
  if (act == 3) return 1.f / (1.f + __expf(-v));     // sigmoid
  return v;
}

__global__ void zero_kernel(float* p, int n) {
  int i = blockIdx.x * blockDim.x + threadIdx.x;
  if (i < n) p[i] = 0.f;
}

// A-fragment from fp32 weights: one contiguous 16-float run, vector loads + cvt.
__device__ __forceinline__ v16bf load_afrag(const float* pa /* row + c0 + 16*half */) {
  v4f a0 = *(const v4f*)(pa + 0);
  v4f a1 = *(const v4f*)(pa + 4);
  v4f a2 = *(const v4f*)(pa + 8);
  v4f a3 = *(const v4f*)(pa + 12);
  v16bf av;
#pragma unroll
  for (int e = 0; e < 4; ++e) {
    av[e]      = (__bf16)a0[e];
    av[e + 4]  = (__bf16)a1[e];
    av[e + 8]  = (__bf16)a2[e];
    av[e + 12] = (__bf16)a3[e];
  }
  return av;
}

// ---------------- GEMM: out[b,o,n] = sum_c W[o,c] * X[b,c,n] ----------------
// Activations read from bf16 transposed mirror Xt[b][n][c] (C = row stride).
template <int C, int O, int NT>
__global__ void conv1d_wmma_kernel(
    const float* __restrict__ Wm, const bf16_t* __restrict__ Xt,
    float* __restrict__ Out, float* __restrict__ stats, const float* __restrict__ bias,
    int Bn, long outBS, int chanOff)
{
  constexpr int Np = 4096;
  constexpr int nT = Np / (16 * NT);
  constexpr int oT = (O + 15) / 16;
  static_assert(C % 32 == 0, "C must be multiple of 32");
  int gwave = (blockIdx.x * blockDim.x + threadIdx.x) >> 5;
  int lane  = threadIdx.x & 31;
  long total = (long)Bn * nT * oT;
  if (gwave >= total) return;
  int bb  = (int)(gwave / (nT * oT));
  int rem = (int)(gwave % (nT * oT));
  int ot = rem / nT, ng = rem % nT;
  int m = lane & 15, half = lane >> 4;
  int oBase = ot * 16;
  int orow  = oBase + m; if (orow > O - 1) orow = O - 1;
  int nBase = ng * (16 * NT);
  const float*  Wrow = Wm + (long)orow * C;
  const bf16_t* Xb   = Xt + (long)bb * Np * C;

  v8f zero8 = {};
  v8f acc[NT];
#pragma unroll
  for (int t = 0; t < NT; ++t) acc[t] = zero8;

  for (int c0 = 0; c0 < C; c0 += 32) {
    v16bf av = load_afrag(Wrow + c0 + 16 * half);
#pragma unroll
    for (int t = 0; t < NT; ++t) {
      const bf16_t* px = Xb + (long)(nBase + t * 16 + m) * C + c0 + 16 * half;
      v16bf bv = *(const v16bf*)px;       // 32B contiguous: 2x global_load_b128
      acc[t] = __builtin_amdgcn_wmma_f32_16x16x32_bf16(false, av, false, bv, (short)0, acc[t], false, false);
    }
  }

#pragma unroll
  for (int t = 0; t < NT; ++t) {
    int nIdx = nBase + t * 16 + m;
#pragma unroll
    for (int r = 0; r < 8; ++r) {
      int oo = oBase + r + 8 * half;
      int oc = oo < O ? oo : O - 1;
      float bsv = bias ? bias[oc] : 0.f;   // clamped, unconditional
      if ((O % 16 == 0) || (oo < O))
        Out[(long)bb * outBS + (long)(chanOff + oo) * Np + nIdx] = acc[t][r] + bsv;
    }
  }
  if (stats) {
#pragma unroll
    for (int r = 0; r < 8; ++r) {
      float s1 = 0.f, s2 = 0.f;
#pragma unroll
      for (int t = 0; t < NT; ++t) { float v = acc[t][r]; s1 += v; s2 += v * v; }
      for (int off = 1; off < 16; off <<= 1) {
        s1 += __shfl_xor(s1, off);
        s2 += __shfl_xor(s2, off);
      }
      int oo = oBase + r + 8 * half;
      if (m == 0 && ((O % 16 == 0) || oo < O)) {
        atomicAdd(&stats[oo], s1);
        atomicAdd(&stats[STATS_CH + oo], s2);
      }
    }
  }
}

// ------- Fused edge-conv: gather + GEMM + max over k + BN stats -------
// out[b,o,n] = max_k sum_c W[o,c] * [F[:,idx[n,k]] ; F[:,n]], F from bf16 [b][n][c] mirror.
template <int Cin, int O, int NT>
__global__ void edgeconv_wmma_kernel(
    const float* __restrict__ Wm, const bf16_t* __restrict__ Ft,
    const int* __restrict__ idx, float* __restrict__ Out, float* __restrict__ stats,
    int Bn, int Kn)
{
  constexpr int Np = 4096;
  constexpr int C2 = 2 * Cin;
  constexpr int nT = Np / (16 * NT);
  constexpr int oT = O / 16;
  static_assert(Cin % 16 == 0 && C2 % 32 == 0, "Cin must be multiple of 16");
  int gwave = (blockIdx.x * blockDim.x + threadIdx.x) >> 5;
  int lane  = threadIdx.x & 31;
  long total = (long)Bn * nT * oT;
  if (gwave >= total) return;
  int bb  = (int)(gwave / (nT * oT));
  int rem = (int)(gwave % (nT * oT));
  int ot = rem / nT, ng = rem % nT;
  int m = lane & 15, half = lane >> 4;
  int oBase = ot * 16;
  int nBase = ng * (16 * NT);
  const float*  Wrow = Wm + (long)(oBase + m) * C2;
  const bf16_t* Fb   = Ft + (long)bb * Np * Cin;

  float maxv[NT][8], s1a[8], s2a[8];
#pragma unroll
  for (int r = 0; r < 8; ++r) { s1a[r] = 0.f; s2a[r] = 0.f; }
#pragma unroll
  for (int t = 0; t < NT; ++t)
#pragma unroll
    for (int r = 0; r < 8; ++r) maxv[t][r] = -3.4e38f;

  for (int k = 0; k < Kn; ++k) {
    int nb[NT];
#pragma unroll
    for (int t = 0; t < NT; ++t)
      nb[t] = idx[((long)bb * Np + (nBase + t * 16 + m)) * Kn + k];
    v8f zero8 = {};
    v8f acc[NT];
#pragma unroll
    for (int t = 0; t < NT; ++t) acc[t] = zero8;

    for (int c0 = 0; c0 < C2; c0 += 32) {
      v16bf av = load_afrag(Wrow + c0 + 16 * half);
      int cBase = c0 + 16 * half;
      bool isNb = cBase < Cin;               // branchless pointer select per lane-half
      int  ch   = isNb ? cBase : cBase - Cin;
#pragma unroll
      for (int t = 0; t < NT; ++t) {
        int col = isNb ? nb[t] : (nBase + t * 16 + m);
        v16bf bv = *(const v16bf*)(Fb + (long)col * Cin + ch);
        acc[t] = __builtin_amdgcn_wmma_f32_16x16x32_bf16(false, av, false, bv, (short)0, acc[t], false, false);
      }
    }
#pragma unroll
    for (int t = 0; t < NT; ++t)
#pragma unroll
      for (int r = 0; r < 8; ++r) {
        float v = acc[t][r];
        maxv[t][r] = fmaxf(maxv[t][r], v);
        s1a[r] += v; s2a[r] += v * v;
      }
  }

#pragma unroll
  for (int t = 0; t < NT; ++t) {
    int nIdx = nBase + t * 16 + m;
#pragma unroll
    for (int r = 0; r < 8; ++r) {
      int oo = oBase + r + 8 * half;
      Out[(long)bb * O * Np + (long)oo * Np + nIdx] = maxv[t][r];
    }
  }
#pragma unroll
  for (int r = 0; r < 8; ++r) {
    float s1 = s1a[r], s2 = s2a[r];
    for (int off = 1; off < 16; off <<= 1) {
      s1 += __shfl_xor(s1, off);
      s2 += __shfl_xor(s2, off);
    }
    int oo = oBase + r + 8 * half;
    if (m == 0) {
      atomicAdd(&stats[oo], s1);
      atomicAdd(&stats[STATS_CH + oo], s2);
    }
  }
}

// ---------------- kNN: pd tile = 2*Gram - xx_n - xx_m via WMMA ----------------
template <int C, int MT>
__global__ void knn_pd_kernel(const bf16_t* __restrict__ Ft, const float* __restrict__ xx,
                              float* __restrict__ pd, int bb)
{
  constexpr int Np = 4096;
  constexpr int t16 = Np / 16;
  constexpr int mT = t16 / MT;
  static_assert(C % 32 == 0, "C must be multiple of 32");
  int gwave = (blockIdx.x * blockDim.x + threadIdx.x) >> 5;
  int lane  = threadIdx.x & 31;
  long total = (long)t16 * mT;
  if (gwave >= total) return;
  int nt = (int)(gwave / mT), mg = (int)(gwave % mT);
  int m = lane & 15, half = lane >> 4;
  int nBase = nt * 16, mBase = mg * (16 * MT);
  const bf16_t* Fb = Ft + (long)bb * Np * C;
  const float*  xb = xx + (long)bb * Np;

  v8f zero8 = {};
  v8f acc[MT];
#pragma unroll
  for (int t = 0; t < MT; ++t) acc[t] = zero8;

  for (int c0 = 0; c0 < C; c0 += 32) {
    v16bf av;
    {
      const bf16_t* pa = Fb + (long)(nBase + m) * C + c0 + 16 * half;
      v8bf a0 = *(const v8bf*)pa;          // 16B vector loads
      v8bf a1 = *(const v8bf*)(pa + 8);
#pragma unroll
      for (int s = 0; s < 8; ++s) { av[s] = a0[s]; av[s + 8] = a1[s]; }
    }
#pragma unroll
    for (int t = 0; t < MT; ++t) {
      const bf16_t* px = Fb + (long)(mBase + t * 16 + m) * C + c0 + 16 * half;
      v16bf bv = *(const v16bf*)px;
      acc[t] = __builtin_amdgcn_wmma_f32_16x16x32_bf16(false, av, false, bv, (short)0, acc[t], false, false);
    }
  }
#pragma unroll
  for (int t = 0; t < MT; ++t) {
    int mCol = mBase + t * 16 + m;
#pragma unroll
    for (int r = 0; r < 8; ++r) {
      int nRow = nBase + r + 8 * half;
      pd[(long)nRow * Np + mCol] = 2.f * acc[t][r] - xb[nRow] - xb[mCol];
    }
  }
}

// one wave per row: per-lane sorted top-20 insertion, then LDS merge
__global__ void topk_kernel(const float* __restrict__ pd, int* __restrict__ idxOut,
                            int bb, int Np, int Kn)
{
  __shared__ float sv[32 * 20];
  __shared__ int   si[32 * 20];
  int row = blockIdx.x;
  int lane = threadIdx.x;
  float bv[20]; int bi[20];
  for (int t = 0; t < Kn; ++t) { bv[t] = -3.4e38f; bi[t] = 0; }
  for (int mcol = lane; mcol < Np; mcol += 32) {
    float v = pd[(long)row * Np + mcol];
    if (v > bv[Kn - 1]) {
      int p = Kn - 1;
      while (p > 0 && bv[p - 1] < v) { bv[p] = bv[p - 1]; bi[p] = bi[p - 1]; --p; }
      bv[p] = v; bi[p] = mcol;
    }
  }
  for (int t = 0; t < Kn; ++t) { sv[lane * 20 + t] = bv[t]; si[lane * 20 + t] = bi[t]; }
  __syncthreads();
  if (lane == 0) {
    int heads[32];
    for (int l = 0; l < 32; ++l) heads[l] = 0;
    for (int t = 0; t < Kn; ++t) {
      float best = -3.4e38f; int bl = 0;
      for (int l = 0; l < 32; ++l) {
        if (heads[l] < Kn) {
          float c = sv[l * 20 + heads[l]];
          if (c > best) { best = c; bl = l; }
        }
      }
      idxOut[((long)bb * Np + row) * Kn + t] = si[bl * 20 + heads[bl]];
      heads[bl]++;
    }
  }
}

// ---------------- elementwise helpers ----------------
// BN + activation in place (f32, channel-major) and optional bf16 transposed mirror.
__global__ void bn_act_kernel(float* __restrict__ data, const float* __restrict__ stats,
                              const float* __restrict__ gamma, const float* __restrict__ beta,
                              float invCnt, int Bn, int O, int Np, long bStride, int chanOff, int act,
                              bf16_t* __restrict__ outT, int Ct, int tOff)
{
  long total = (long)Bn * O * Np;
  long i = (long)blockIdx.x * blockDim.x + threadIdx.x;
  if (i >= total) return;
  int n = (int)(i % Np);
  long t = i / Np;
  int o = (int)(t % O);
  int b = (int)(t / O);
  float mean = stats[o] * invCnt;
  float var  = stats[STATS_CH + o] * invCnt - mean * mean;
  float sc = gamma[o] * rsqrtf(var + EPS);
  long off = (long)b * bStride + (long)(chanOff + o) * Np + n;
  float v = act_apply(sc * (data[off] - mean) + beta[o], act);
  data[off] = v;
  if (outT) outT[((long)b * Np + n) * Ct + tOff + o] = (bf16_t)v;
}

__global__ void xxt_kernel(const bf16_t* __restrict__ Ft, float* __restrict__ xx,
                           int Bn, int C, int Np) {
  long i = (long)blockIdx.x * blockDim.x + threadIdx.x;
  if (i >= (long)Bn * Np) return;
  float s = 0.f;
  const bf16_t* p = Ft + i * C;
  for (int c = 0; c < C; ++c) {
    float v = (float)p[c];
    s += v * v;
  }
  xx[i] = s;
}

__global__ void chan_reduce_kernel(const float* __restrict__ in, float* __restrict__ out,
                                   int Np, int mode) { // 0=max, 1=mean
  __shared__ float sh[256];
  long bc = blockIdx.x;
  float acc = (mode == 0) ? -3.4e38f : 0.f;
  for (int n = threadIdx.x; n < Np; n += blockDim.x) {
    float v = in[bc * Np + n];
    acc = (mode == 0) ? fmaxf(acc, v) : acc + v;
  }
  sh[threadIdx.x] = acc;
  __syncthreads();
  for (int s = 128; s > 0; s >>= 1) {
    if ((int)threadIdx.x < s)
      sh[threadIdx.x] = (mode == 0) ? fmaxf(sh[threadIdx.x], sh[threadIdx.x + s])
                                    : sh[threadIdx.x] + sh[threadIdx.x + s];
    __syncthreads();
  }
  if (threadIdx.x == 0) out[bc] = (mode == 0) ? sh[0] : sh[0] / Np;
}

__global__ void fc_kernel(const float* __restrict__ W, const float* __restrict__ bias,
                          const float* __restrict__ in, float* __restrict__ out,
                          int Bn, int Cin, int Cout, int act) {
  int i = blockIdx.x * blockDim.x + threadIdx.x;
  if (i >= Bn * Cout) return;
  int b = i / Cout, o = i % Cout;
  float s = bias ? bias[o] : 0.f;
  for (int c = 0; c < Cin; ++c) s += in[b * Cin + c] * W[(long)o * Cin + c];
  out[i] = act_apply(s, act);
}

// writes bf16 transposed (pre-zeroed) tensor rows d<3
__global__ void transform_kernel(const float* __restrict__ xin, long xBS,
                                 const float* __restrict__ m9, bf16_t* __restrict__ outT,
                                 int Bn, int Np, int Ct) {
  long i = (long)blockIdx.x * blockDim.x + threadIdx.x;
  if (i >= (long)Bn * 3 * Np) return;
  int n = (int)(i % Np);
  int d = (int)((i / Np) % 3);
  int b = (int)(i / ((long)3 * Np));
  float s = 0.f;
  for (int c = 0; c < 3; ++c)
    s += xin[(long)b * xBS + (long)c * Np + n] * (m9[b * 9 + c * 3 + d] + (c == d ? 1.f : 0.f));
  outT[((long)b * Np + n) * Ct + d] = (bf16_t)s;
}

__global__ void pad_xt_kernel(const float* __restrict__ X, bf16_t* __restrict__ Xt,
                              int Bn, int c0, int Np) { // -> [b][n][32], rows 3.. zero
  long total = (long)Bn * Np * 32;
  long i = (long)blockIdx.x * blockDim.x + threadIdx.x;
  if (i >= total) return;
  int c = (int)(i % 32);
  int n = (int)((i / 32) % Np);
  int b = (int)(i / ((long)32 * Np));
  float v = (c < 3) ? X[(long)b * 6 * Np + (long)(c0 + c) * Np + n] : 0.f;
  Xt[i] = (bf16_t)v;
}

__global__ void pad_conv_w(const float* __restrict__ W, float* __restrict__ Wp,
                           int O, int Cin, int CinPad) {
  int i = blockIdx.x * blockDim.x + threadIdx.x;
  if (i >= O * CinPad) return;
  int o = i / CinPad, c = i % CinPad;
  Wp[i] = (c < Cin) ? W[o * Cin + c] : 0.f;
}

__global__ void pad_edge_w(const float* __restrict__ W, float* __restrict__ Wp,
                           int O, int Cin, int CinPad) { // (O,2*Cin) -> (O,2*CinPad)
  int i = blockIdx.x * blockDim.x + threadIdx.x;
  if (i >= O * 2 * CinPad) return;
  int o = i / (2 * CinPad), c = i % (2 * CinPad);
  float v = 0.f;
  if (c < Cin)                              v = W[o * 2 * Cin + c];
  else if (c >= CinPad && c < CinPad + Cin) v = W[o * 2 * Cin + Cin + (c - CinPad)];
  Wp[i] = v;
}

// concat 3 channel-major f32 tensors into bf16 transposed [b][n][Ca+Cb+Cc]
__global__ void concat3_t_kernel(const float* __restrict__ a, int Ca,
                                 const float* __restrict__ b_, int Cb,
                                 const float* __restrict__ c_, int Cc,
                                 bf16_t* __restrict__ outT, int Bn, int Np) {
  int Ct = Ca + Cb + Cc;
  long total = (long)Bn * Np * Ct;
  long i = (long)blockIdx.x * blockDim.x + threadIdx.x;
  if (i >= total) return;
  int o = (int)(i % Ct);
  int n = (int)((i / Ct) % Np);
  int b = (int)(i / ((long)Ct * Np));
  float v;
  if (o < Ca)           v = a [((long)b * Ca + o) * Np + n];
  else if (o < Ca + Cb) v = b_[((long)b * Cb + (o - Ca)) * Np + n];
  else                  v = c_[((long)b * Cc + (o - Ca - Cb)) * Np + n];
  outT[i] = (bf16_t)v;
}

__global__ void scale_kernel(float* __restrict__ f, bf16_t* __restrict__ ft,
                             const float* __restrict__ w, int Bn, int C, int Np) {
  long total = (long)Bn * C * Np;
  long i = (long)blockIdx.x * blockDim.x + threadIdx.x;
  if (i >= total) return;
  int n = (int)(i % Np);
  int c = (int)((i / Np) % C);
  int b = (int)(i / ((long)C * Np));
  float v = f[i] * w[b * C + c];
  f[i] = v;
  ft[((long)b * Np + n) * C + c] = (bf16_t)v;
}

// ------------------------------------------------------------------
static inline int eb(long n, long t) { return (int)((n + t - 1) / t); }
static const int B_ = 4, N_ = 4096, KK_ = 20;

template <int C, int O>
static void run_conv(const float* Wm, const bf16_t* Xt, float* Outp, float* st,
                     const float* bias, long oBS, int cOff, hipStream_t stream) {
  constexpr int NT = 4;
  long jobs = (long)B_ * (N_ / (16 * NT)) * ((O + 15) / 16);
  conv1d_wmma_kernel<C, O, NT><<<eb(jobs, 8), 256, 0, stream>>>(
      Wm, Xt, Outp, st, bias, B_, oBS, cOff);
}

template <int C, int O>
static void graph_round(const bf16_t* cint, const bf16_t* nint,
                        const float* wc, const float* gc, const float* bc,
                        const float* wn, const float* gn, const float* bnn,
                        float* cout_, bf16_t* cout_t, float* nout_, bf16_t* nout_t,
                        float* xxb, float* pd, int* idx, float* stats,
                        hipStream_t stream) {
  xxt_kernel<<<eb((long)B_ * N_, 256), 256, 0, stream>>>(cint, xxb, B_, C, N_);
  for (int b = 0; b < B_; ++b) {
    constexpr int MT = 4;
    long jobs = (long)(N_ / 16) * (N_ / (16 * MT));
    knn_pd_kernel<C, MT><<<eb(jobs, 8), 256, 0, stream>>>(cint, xxb, pd, b);
    topk_kernel<<<N_, 32, 0, stream>>>(pd, idx, b, N_, KK_);
  }
  constexpr int NT = 2;
  long ejobs = (long)B_ * (N_ / (16 * NT)) * (O / 16);
  float invC = 1.f / ((float)B_ * N_ * KK_);
  zero_kernel<<<eb(2 * STATS_CH, 256), 256, 0, stream>>>(stats, 2 * STATS_CH);
  edgeconv_wmma_kernel<C, O, NT><<<eb(ejobs, 8), 256, 0, stream>>>(wc, cint, idx, cout_, stats, B_, KK_);
  bn_act_kernel<<<eb((long)B_ * O * N_, 256), 256, 0, stream>>>(
      cout_, stats, gc, bc, invC, B_, O, N_, (long)O * N_, 0, 2, cout_t, O, 0);
  zero_kernel<<<eb(2 * STATS_CH, 256), 256, 0, stream>>>(stats, 2 * STATS_CH);
  edgeconv_wmma_kernel<C, O, NT><<<eb(ejobs, 8), 256, 0, stream>>>(wn, nint, idx, nout_, stats, B_, KK_);
  bn_act_kernel<<<eb((long)B_ * O * N_, 256), 256, 0, stream>>>(
      nout_, stats, gn, bnn, invC, B_, O, N_, (long)O * N_, 0, 2, nout_t, O, 0);
}

extern "C" void kernel_launch(void* const* d_in, const int* in_sizes, int n_in,
                              void* d_out, int out_size, void* d_ws, size_t ws_size,
                              hipStream_t stream) {
  (void)in_sizes; (void)n_in; (void)out_size; (void)ws_size;
  const float* X = (const float*)d_in[0];
  auto P = [&](int i) { return (const float*)d_in[i]; };

  float* ws = (float*)d_ws;
  long off = 0;
  auto allocf = [&](long n) { float* p = ws + off; off += (n + 15) & ~15L; return p; };
  auto allocb = [&](long n) { bf16_t* p = (bf16_t*)(ws + off); off += ((n + 1) / 2 + 15) & ~15L; return p; };

  // fp32 channel-major buffers
  float* h1 = allocf((long)B_ * 64 * N_);
  float* h2 = allocf((long)B_ * 128 * N_);
  float* h3 = allocf((long)B_ * 1024 * N_);
  float* pool = allocf(B_ * 1024);
  float* fca = allocf(B_ * 512);
  float* fcb = allocf(B_ * 256);
  float* mat = allocf(B_ * 9);
  float* stats = allocf(2 * STATS_CH);
  float* wp1 = allocf(64 * 32);
  float* ewc = allocf(64 * 64);
  float* ewn = allocf(64 * 64);
  float* xxb = allocf((long)B_ * N_);
  float* pd  = allocf((long)N_ * N_);
  int*   idx = (int*)allocf((long)B_ * N_ * KK_);
  float* c1 = allocf((long)B_ * 64 * N_);  float* n1 = allocf((long)B_ * 64 * N_);
  float* c2 = allocf((long)B_ * 128 * N_); float* n2 = allocf((long)B_ * 128 * N_);
  float* c3 = allocf((long)B_ * 256 * N_); float* n3 = allocf((long)B_ * 256 * N_);
  float* fbuf = allocf((long)B_ * 512 * N_);
  float* gmean = allocf(B_ * 512);
  float* att1  = allocf(B_ * 256);
  float* attw  = allocf(B_ * 512);
  float* hp1 = allocf((long)B_ * 512 * N_);
  float* hp2 = allocf((long)B_ * 256 * N_);
  float* hp3 = allocf((long)B_ * 128 * N_);
  // bf16 transposed mirrors [b][n][c]
  bf16_t* xcp_t   = allocb((long)B_ * N_ * 32);
  bf16_t* xnp_t   = allocb((long)B_ * N_ * 32);
  bf16_t* coorT_t = allocb((long)B_ * N_ * 32);
  bf16_t* norT_t  = allocb((long)B_ * N_ * 32);
  bf16_t* h1_t = allocb((long)B_ * N_ * 64);
  bf16_t* h2_t = allocb((long)B_ * N_ * 128);
  bf16_t* c1_t = allocb((long)B_ * N_ * 64);  bf16_t* n1_t = allocb((long)B_ * N_ * 64);
  bf16_t* c2_t = allocb((long)B_ * N_ * 128); bf16_t* n2_t = allocb((long)B_ * N_ * 128);
  bf16_t* c3_t = allocb((long)B_ * N_ * 256); bf16_t* n3_t = allocb((long)B_ * N_ * 256);
  bf16_t* cat_t  = allocb((long)B_ * N_ * 448);
  bf16_t* fbuf_t = allocb((long)B_ * N_ * 512);
  bf16_t* hp1_t = allocb((long)B_ * N_ * 512);
  bf16_t* hp2_t = allocb((long)B_ * N_ * 256);
  bf16_t* hp3_t = allocb((long)B_ * N_ * 128);

  auto zstats = [&] {
    zero_kernel<<<eb(2 * STATS_CH, 256), 256, 0, stream>>>(stats, 2 * STATS_CH);
  };
  auto bn = [&](float* data, const float* g, const float* b, float invCnt,
                int O, long bS, int cOff, int act, bf16_t* outT, int Ct, int tOff) {
    bn_act_kernel<<<eb((long)B_ * O * N_, 256), 256, 0, stream>>>(
        data, stats, g, b, invCnt, B_, O, N_, bS, cOff, act, outT, Ct, tOff);
  };
  float invBN = 1.f / ((float)B_ * N_);

  // ---- pad raw inputs to bf16 [b][n][32] ----
  pad_xt_kernel<<<eb((long)B_ * N_ * 32, 256), 256, 0, stream>>>(X, xcp_t, B_, 0, N_);
  pad_xt_kernel<<<eb((long)B_ * N_ * 32, 256), 256, 0, stream>>>(X, xnp_t, B_, 3, N_);
  zero_kernel<<<eb((long)B_ * N_ * 16, 256), 256, 0, stream>>>((float*)coorT_t, B_ * N_ * 16);
  zero_kernel<<<eb((long)B_ * N_ * 16, 256), 256, 0, stream>>>((float*)norT_t, B_ * N_ * 16);

  // ---- STN (coor then nor) ----
  auto stn = [&](const bf16_t* xpad_t, const float* rawx, int qi, bf16_t* outT) {
    pad_conv_w<<<eb(64 * 32, 256), 256, 0, stream>>>(P(qi + 0), wp1, 64, 3, 32);
    zstats(); run_conv<32, 64>(wp1, xpad_t, h1, stats, nullptr, (long)64 * N_, 0, stream);
    bn(h1, P(qi + 1), P(qi + 2), invBN, 64, (long)64 * N_, 0, 1, h1_t, 64, 0);
    zstats(); run_conv<64, 128>(P(qi + 3), h1_t, h2, stats, nullptr, (long)128 * N_, 0, stream);
    bn(h2, P(qi + 4), P(qi + 5), invBN, 128, (long)128 * N_, 0, 1, h2_t, 128, 0);
    zstats(); run_conv<128, 1024>(P(qi + 6), h2_t, h3, stats, nullptr, (long)1024 * N_, 0, stream);
    bn(h3, P(qi + 7), P(qi + 8), invBN, 1024, (long)1024 * N_, 0, 1, nullptr, 0, 0);
    chan_reduce_kernel<<<B_ * 1024, 256, 0, stream>>>(h3, pool, N_, 0);
    fc_kernel<<<eb(B_ * 512, 256), 256, 0, stream>>>(P(qi + 9),  P(qi + 10), pool, fca, B_, 1024, 512, 1);
    fc_kernel<<<eb(B_ * 256, 256), 256, 0, stream>>>(P(qi + 11), P(qi + 12), fca,  fcb, B_, 512, 256, 1);
    fc_kernel<<<eb(B_ * 9,   256), 256, 0, stream>>>(P(qi + 13), P(qi + 14), fcb,  mat, B_, 256, 9, 0);
    transform_kernel<<<eb((long)B_ * 3 * N_, 256), 256, 0, stream>>>(rawx, (long)6 * N_, mat, outT, B_, N_, 32);
  };
  stn(xcp_t, X,          1,  coorT_t);
  stn(xnp_t, X + 3 * N_, 16, norT_t);

  // ---- three graph-feature / edge-conv rounds ----
  pad_edge_w<<<eb(64 * 64, 256), 256, 0, stream>>>(P(31), ewc, 64, 3, 32);
  pad_edge_w<<<eb(64 * 64, 256), 256, 0, stream>>>(P(43), ewn, 64, 3, 32);
  graph_round<32, 64>(coorT_t, norT_t, ewc, P(32), P(33), ewn, P(44), P(45),
                      c1, c1_t, n1, n1_t, xxb, pd, idx, stats, stream);
  graph_round<64, 128>(c1_t, n1_t, P(34), P(35), P(36), P(46), P(47), P(48),
                       c2, c2_t, n2, n2_t, xxb, pd, idx, stats, stream);
  graph_round<128, 256>(c2_t, n2_t, P(37), P(38), P(39), P(49), P(50), P(51),
                        c3, c3_t, n3, n3_t, xxb, pd, idx, stats, stream);

  // ---- c5 / n5 -> f (B,512,N) ----
  concat3_t_kernel<<<eb((long)B_ * N_ * 448, 256), 256, 0, stream>>>(c1, 64, c2, 128, c3, 256, cat_t, B_, N_);
  zstats(); run_conv<448, 256>(P(40), cat_t, fbuf, stats, nullptr, (long)512 * N_, 0, stream);
  bn(fbuf, P(41), P(42), invBN, 256, (long)512 * N_, 0, 2, fbuf_t, 512, 0);
  concat3_t_kernel<<<eb((long)B_ * N_ * 448, 256), 256, 0, stream>>>(n1, 64, n2, 128, n3, 256, cat_t, B_, N_);
  zstats(); run_conv<448, 256>(P(52), cat_t, fbuf, stats, nullptr, (long)512 * N_, 256, stream);
  bn(fbuf, P(53), P(54), invBN, 256, (long)512 * N_, 256, 2, fbuf_t, 512, 256);

  // ---- channel attention ----
  chan_reduce_kernel<<<B_ * 512, 256, 0, stream>>>(fbuf, gmean, N_, 1);
  fc_kernel<<<eb(B_ * 256, 256), 256, 0, stream>>>(P(55), P(56), gmean, att1, B_, 512, 256, 1);
  fc_kernel<<<eb(B_ * 512, 256), 256, 0, stream>>>(P(57), P(58), att1, attw, B_, 256, 512, 3);
  scale_kernel<<<eb((long)B_ * 512 * N_, 256), 256, 0, stream>>>(fbuf, fbuf_t, attw, B_, 512, N_);

  // ---- head ----
  zstats(); run_conv<512, 512>(P(59), fbuf_t, hp1, stats, nullptr, (long)512 * N_, 0, stream);
  bn(hp1, P(60), P(61), invBN, 512, (long)512 * N_, 0, 2, hp1_t, 512, 0);
  zstats(); run_conv<512, 256>(P(62), hp1_t, hp2, stats, nullptr, (long)256 * N_, 0, stream);
  bn(hp2, P(63), P(64), invBN, 256, (long)256 * N_, 0, 2, hp2_t, 256, 0);
  zstats(); run_conv<256, 128>(P(65), hp2_t, hp3, stats, nullptr, (long)128 * N_, 0, stream);
  bn(hp3, P(66), P(67), invBN, 128, (long)128 * N_, 0, 2, hp3_t, 128, 0);
  run_conv<128, 15>(P(68), hp3_t, (float*)d_out, nullptr, P(69), (long)15 * N_, 0, stream);
}
